// SPGANGenerator_2439541424689
// MI455X (gfx1250) — matrix operand order repaired and checked by reading, not compiled, not run on hardware
//
#include <hip/hip_runtime.h>
#include <hip/hip_bf16.h>

typedef __attribute__((ext_vector_type(16))) _Float16 v16h;
typedef __attribute__((ext_vector_type(8)))  float    v8f;

#define BB 16
#define NN 2048
#define KK 10
#define EPS 1e-5f

// ---------------------------------------------------------------------------
// Fragment loaders for V_WMMA_F32_16X16X32_F16 (CDNA5 layouts, CDNA5 ISA)
// A: 16x32 (MxK) f16. lanes0-15: M=lane, K-groups {0..7,16..23}; lanes16-31: +8.
// ---------------------------------------------------------------------------
__device__ inline v16h load_a_frag(const float* a /*16x32 row-major*/, int lane) {
    v16h f;
    int m  = lane & 15;
    int hi = (lane >> 4) & 1;
#pragma unroll
    for (int v = 0; v < 8; ++v) {
        int kb = (v < 4 ? 2 * v : 16 + 2 * (v - 4)) + hi * 8;
        f[2 * v]     = (_Float16)a[m * 32 + kb];
        f[2 * v + 1] = (_Float16)a[m * 32 + kb + 1];
    }
    return f;
}
// B: 32x16 (KxN) f16. VGPR v, half h, lane l: N=l&15, K=(v<4?2v:16+2(v-4))+h+(l<16?0:8).
// Tile stored as [n(16)][k(32)] row-major.
__device__ inline v16h load_b_frag(const float* w, int lane) {
    v16h f;
    int n  = lane & 15;
    int hi = (lane >> 4) & 1;
#pragma unroll
    for (int v = 0; v < 8; ++v) {
        int k0 = (v < 4 ? 2 * v : 16 + 2 * (v - 4)) + hi * 8;
        f[2 * v]     = (_Float16)w[n * 32 + k0];
        f[2 * v + 1] = (_Float16)w[n * 32 + k0 + 1];
    }
    return f;
}

// ---------------------------------------------------------------------------
// Generic GEMM: Out[M x O] = act(A[M x Kc](lda) * W[O x Kc]^T + bias)
// Block tile 64x64: 4 waves, each wave = 16 rows x 64 cols (4 accumulators,
// 4 WMMAs per k-chunk reusing one A fragment). Full interior tiles are staged
// global->LDS with CDNA5 async loads (ASYNCcnt + s_wait_asynccnt).
// ---------------------------------------------------------------------------
__global__ void __launch_bounds__(128) gemm_wmma(
        const float* __restrict__ A, int lda, int M, int Kc,
        const float* __restrict__ W, int ldw,
        const float* __restrict__ bias,
        float* __restrict__ Out, int ldo, int O, float slope) {
    __shared__ float As[64 * 32];
    __shared__ float Ws[64 * 32];
    const int tid  = threadIdx.x;
    const int lane = tid & 31;
    const int wv   = tid >> 5;
    const int m0   = blockIdx.x * 64;
    const int n0   = blockIdx.y * 64;

    v8f acc0 = {}, acc1 = {}, acc2 = {}, acc3 = {};
    for (int kc0 = 0; kc0 < Kc; kc0 += 32) {
        const bool full = (kc0 + 32 <= Kc) && (m0 + 64 <= M) && (n0 + 64 <= O);
        if (full) {
            // gfx1250 async global->LDS staging (bypasses VGPRs; ASYNCcnt)
            for (int e = tid; e < 2048; e += 128) {
                int r = e >> 5, c = e & 31;
                const float* ga = A + (size_t)(m0 + r) * lda + (kc0 + c);
                const float* gw = W + (size_t)(n0 + r) * ldw + (kc0 + c);
                unsigned la = (unsigned)(size_t)(&As[e]);
                unsigned lw = (unsigned)(size_t)(&Ws[e]);
                asm volatile("global_load_async_to_lds_b32 %0, %1, off"
                             :: "v"(la), "v"(ga) : "memory");
                asm volatile("global_load_async_to_lds_b32 %0, %1, off"
                             :: "v"(lw), "v"(gw) : "memory");
            }
            asm volatile("s_wait_asynccnt 0x0" ::: "memory");
        } else {
            for (int e = tid; e < 2048; e += 128) {
                int r = e >> 5, c = e & 31, k = kc0 + c;
                int mr = m0 + r, orow = n0 + r;
                As[e] = (mr < M && k < Kc) ? A[(size_t)mr * lda + k] : 0.f;
                Ws[e] = (orow < O && k < Kc) ? W[(size_t)orow * ldw + k] : 0.f;
            }
        }
        __syncthreads();
        v16h af = load_a_frag(&As[wv * 512], lane);
        v16h b0 = load_b_frag(&Ws[0], lane);
        v16h b1 = load_b_frag(&Ws[512], lane);
        v16h b2 = load_b_frag(&Ws[1024], lane);
        v16h b3 = load_b_frag(&Ws[1536], lane);
        acc0 = __builtin_amdgcn_wmma_f32_16x16x32_f16(false, af, false, b0, (short)0, acc0, false, false);
        acc1 = __builtin_amdgcn_wmma_f32_16x16x32_f16(false, af, false, b1, (short)0, acc1, false, false);
        acc2 = __builtin_amdgcn_wmma_f32_16x16x32_f16(false, af, false, b2, (short)0, acc2, false, false);
        acc3 = __builtin_amdgcn_wmma_f32_16x16x32_f16(false, af, false, b3, (short)0, acc3, false, false);
        __syncthreads();
    }
    // D layout: VGPR r -> M = r + (lane<16?0:8), N = lane&15
    const int nloc  = lane & 15;
    const int mbase = m0 + wv * 16 + ((lane >> 4) & 1) * 8;
#pragma unroll
    for (int t = 0; t < 4; ++t) {
        v8f acc = (t == 0) ? acc0 : (t == 1) ? acc1 : (t == 2) ? acc2 : acc3;
        int nn = n0 + t * 16 + nloc;
        if (nn < O) {
            float bv = bias ? bias[nn] : 0.f;
#pragma unroll
            for (int r = 0; r < 8; ++r) {
                int m = mbase + r;
                if (m < M) {
                    float v = acc[r] + bv;
                    v = (v >= 0.f) ? v : slope * v;
                    Out[(size_t)m * ldo + nn] = v;
                }
            }
        }
    }
}

// ---------------------------------------------------------------------------
// kNN: per point, brute-force top-10 (excluding self), candidates tiled in LDS
// ---------------------------------------------------------------------------
template <int C, int CH>
__global__ void knn_kernel(const float* __restrict__ X /* [B][N][C] */,
                           int* __restrict__ idx /* [B][N][K] */) {
    __shared__ float tile[CH * C];
    __shared__ float tsq[CH];
    const int tid = threadIdx.x;
    const int bpb = NN / 256;
    const int b   = blockIdx.x / bpb;
    const int n   = (blockIdx.x % bpb) * 256 + tid;
    const float* xb = X + (size_t)b * NN * C;

    float p[C];
    float sqo = 0.f;
#pragma unroll
    for (int c = 0; c < C; ++c) { p[c] = xb[(size_t)n * C + c]; sqo += p[c] * p[c]; }

    float bd[KK]; int bi[KK];
#pragma unroll
    for (int j = 0; j < KK; ++j) { bd[j] = 3.4e38f; bi[j] = 0; }

    for (int m0 = 0; m0 < NN; m0 += CH) {
        for (int e = tid; e < CH * C; e += 256) tile[e] = xb[(size_t)m0 * C + e];
        __syncthreads();
        for (int r = tid; r < CH; r += 256) {
            float s = 0.f;
#pragma unroll
            for (int c = 0; c < C; ++c) { float v = tile[r * C + c]; s += v * v; }
            tsq[r] = s;
        }
        __syncthreads();
        if (m0 + CH < NN) __builtin_prefetch(xb + (size_t)(m0 + CH) * C, 0, 1);
        for (int mm = 0; mm < CH; ++mm) {
            int m = m0 + mm;
            if (m == n) continue;
            float dot = 0.f;
#pragma unroll
            for (int c = 0; c < C; ++c) dot += p[c] * tile[mm * C + c];
            float d = sqo + tsq[mm] - 2.f * dot;
            if (d < bd[KK - 1]) {          // bubble-insert, fully unrolled (registers)
                float cd = d; int ci = m;
#pragma unroll
                for (int q = 0; q < KK; ++q) {
                    if (cd < bd[q]) {
                        float td = bd[q]; int ti = bi[q];
                        bd[q] = cd; bi[q] = ci; cd = td; ci = ti;
                    }
                }
            }
        }
        __syncthreads();
    }
#pragma unroll
    for (int j = 0; j < KK; ++j) idx[((size_t)b * NN + n) * KK + j] = bi[j];
}

// ---------------------------------------------------------------------------
// Edge feature gather: ee[(bn*K+kk)][c] = c<C ? x[bn][c] : x[bj][c-C]-x[bn][c-C]
// ---------------------------------------------------------------------------
__global__ void build_ee(const float* __restrict__ X, const int* __restrict__ idx,
                         float* __restrict__ ee, int C, size_t total) {
    int twoC = 2 * C;
    for (size_t i = (size_t)blockIdx.x * blockDim.x + threadIdx.x; i < total;
         i += (size_t)gridDim.x * blockDim.x) {
        size_t r = i / twoC; int c = (int)(i % twoC);
        size_t bn = r / KK; int kk = (int)(r % KK);
        float v;
        if (c < C) v = X[bn * C + c];
        else {
            int j = idx[bn * KK + kk];
            size_t b = bn / NN;
            v = X[((size_t)b * NN + j) * C + (c - C)] - X[bn * C + (c - C)];
        }
        ee[i] = v;
    }
}

// per-channel mean/var over M rows (layout [r][C])
__global__ void colstats(const float* __restrict__ A, int C, size_t M,
                         float* __restrict__ mean, float* __restrict__ var) {
    __shared__ float ss[256], sq[256];
    int c = blockIdx.x;
    float s = 0.f, q = 0.f;
    for (size_t r = threadIdx.x; r < M; r += 256) {
        float v = A[r * C + c]; s += v; q += v * v;
    }
    ss[threadIdx.x] = s; sq[threadIdx.x] = q; __syncthreads();
    for (int st = 128; st > 0; st >>= 1) {
        if ((int)threadIdx.x < st) { ss[threadIdx.x] += ss[threadIdx.x + st]; sq[threadIdx.x] += sq[threadIdx.x + st]; }
        __syncthreads();
    }
    if (threadIdx.x == 0) {
        float m = ss[0] / (float)M;
        mean[c] = m; var[c] = sq[0] / (float)M - m * m;
    }
}

__global__ void bn_apply(float* __restrict__ A, int C, size_t total,
                         const float* __restrict__ mean, const float* __restrict__ var,
                         float slope) {
    for (size_t i = (size_t)blockIdx.x * blockDim.x + threadIdx.x; i < total;
         i += (size_t)gridDim.x * blockDim.x) {
        int c = (int)(i % C);
        float v = (A[i] - mean[c]) * rsqrtf(var[c] + EPS);
        A[i] = (v >= 0.f) ? v : slope * v;
    }
}

// softmax over k (elements strided by C within a (b,n) group)
__global__ void softmax_k(float* __restrict__ A, int C, size_t totalBNC) {
    for (size_t i = (size_t)blockIdx.x * blockDim.x + threadIdx.x; i < totalBNC;
         i += (size_t)gridDim.x * blockDim.x) {
        size_t bn = i / C; int c = (int)(i % C);
        size_t base = bn * KK * (size_t)C + c;
        float mx = -3.4e38f;
#pragma unroll
        for (int k = 0; k < KK; ++k) mx = fmaxf(mx, A[base + (size_t)k * C]);
        float s = 0.f;
#pragma unroll
        for (int k = 0; k < KK; ++k) {
            float e = __expf(A[base + (size_t)k * C] - mx);
            A[base + (size_t)k * C] = e; s += e;
        }
        float inv = 1.f / s;
#pragma unroll
        for (int k = 0; k < KK; ++k) A[base + (size_t)k * C] *= inv;
    }
}

__global__ void mul_inplace(float* __restrict__ a, const float* __restrict__ b, size_t total) {
    for (size_t i = (size_t)blockIdx.x * blockDim.x + threadIdx.x; i < total;
         i += (size_t)gridDim.x * blockDim.x) a[i] *= b[i];
}

// wo[o][c][k] -> wt[o][k*C + c]
__global__ void transpose_wo(const float* __restrict__ wo, float* __restrict__ wt,
                             int O, int C) {
    size_t total = (size_t)O * C * KK;
    for (size_t i = (size_t)blockIdx.x * blockDim.x + threadIdx.x; i < total;
         i += (size_t)gridDim.x * blockDim.x) {
        int o = (int)(i / (C * KK)); int rem = (int)(i % (C * KK));
        int c = rem / KK, k = rem % KK;
        wt[(size_t)o * C * KK + k * C + c] = wo[i];
    }
}

// instance-norm stats: per (b,c) over n
__global__ void inorm_stats(const float* __restrict__ X, int C,
                            float* __restrict__ mean, float* __restrict__ var) {
    __shared__ float ss[256], sq[256];
    int b = blockIdx.x / C, c = blockIdx.x % C;
    float s = 0.f, q = 0.f;
    for (int n = threadIdx.x; n < NN; n += 256) {
        float v = X[((size_t)b * NN + n) * C + c]; s += v; q += v * v;
    }
    ss[threadIdx.x] = s; sq[threadIdx.x] = q; __syncthreads();
    for (int st = 128; st > 0; st >>= 1) {
        if ((int)threadIdx.x < st) { ss[threadIdx.x] += ss[threadIdx.x + st]; sq[threadIdx.x] += sq[threadIdx.x + st]; }
        __syncthreads();
    }
    if (threadIdx.x == 0) {
        float m = ss[0] / (float)NN;
        mean[blockIdx.x] = m; var[blockIdx.x] = sq[0] / (float)NN - m * m;
    }
}

// X[bn][c] = g * inorm(X) + bt   with s[bn][2C] = [g | bt]
__global__ void adain_apply(float* __restrict__ X, const float* __restrict__ s, int C,
                            const float* __restrict__ mean, const float* __restrict__ var) {
    size_t total = (size_t)BB * NN * C;
    for (size_t i = (size_t)blockIdx.x * blockDim.x + threadIdx.x; i < total;
         i += (size_t)gridDim.x * blockDim.x) {
        size_t r = i / C; int c = (int)(i % C);
        int b = (int)(r / NN);
        float g  = s[r * 2 * C + c];
        float bt = s[r * 2 * C + C + c];
        float m = mean[b * C + c], v = var[b * C + c];
        X[i] = g * (X[i] - m) * rsqrtf(v + EPS) + bt;
    }
}

__global__ void rowmax(const float* __restrict__ X, int C, float* __restrict__ out) {
    int i = blockIdx.x * blockDim.x + threadIdx.x;
    if (i >= BB * C) return;
    int b = i / C, c = i % C;
    float m = -3.4e38f;
    for (int n = 0; n < NN; ++n) m = fmaxf(m, X[((size_t)b * NN + n) * C + c]);
    out[i] = m;
}

// bn over batch (16 rows) per column, then lrelu, in place
__global__ void bn1_apply(float* __restrict__ A, int C, float slope) {
    int c = blockIdx.x * blockDim.x + threadIdx.x;
    if (c >= C) return;
    float s = 0.f, q = 0.f;
    for (int b = 0; b < BB; ++b) { float v = A[b * C + c]; s += v; q += v * v; }
    float m = s / (float)BB, v = q / (float)BB - m * m, inv = rsqrtf(v + EPS);
    for (int b = 0; b < BB; ++b) {
        float t = (A[b * C + c] - m) * inv;
        A[b * C + c] = (t >= 0.f) ? t : slope * t;
    }
}

__global__ void build_style_in(const float* __restrict__ x, const float* __restrict__ z,
                               float* __restrict__ out) {
    size_t total = (size_t)BB * NN * 131;
    for (size_t i = (size_t)blockIdx.x * blockDim.x + threadIdx.x; i < total;
         i += (size_t)gridDim.x * blockDim.x) {
        size_t r = i / 131; int c = (int)(i % 131);
        out[i] = (c < 3) ? x[r * 3 + c] : z[r * 128 + (c - 3)];
    }
}

__global__ void build_feat(const float* __restrict__ fgf /*16x512*/,
                           const float* __restrict__ x2 /*[B*N][128]*/,
                           float* __restrict__ out) {
    size_t total = (size_t)BB * NN * 640;
    for (size_t i = (size_t)blockIdx.x * blockDim.x + threadIdx.x; i < total;
         i += (size_t)gridDim.x * blockDim.x) {
        size_t r = i / 640; int c = (int)(i % 640);
        int b = (int)(r / NN);
        out[i] = (c < 512) ? fgf[b * 512 + c] : x2[r * 128 + (c - 512)];
    }
}

// ---------------------------------------------------------------------------
extern "C" void kernel_launch(void* const* d_in, const int* in_sizes, int n_in,
                              void* d_out, int out_size, void* d_ws, size_t ws_size,
                              hipStream_t stream) {
    const float* z    = (const float*)d_in[0];
    const float* x    = (const float*)d_in[1];
    const float* hw1  = (const float*)d_in[2];
    const float* hb1  = (const float*)d_in[3];
    const float* hw2  = (const float*)d_in[4];
    const float* hb2  = (const float*)d_in[5];
    const float* e1w1 = (const float*)d_in[6];
    const float* e1b1 = (const float*)d_in[7];
    const float* e1w2 = (const float*)d_in[8];
    const float* e1b2 = (const float*)d_in[9];
    const float* e1wx = (const float*)d_in[10];
    const float* e1bx = (const float*)d_in[11];
    const float* e1wo = (const float*)d_in[12];
    const float* e1bo = (const float*)d_in[13];
    const float* a1w  = (const float*)d_in[14];
    const float* a1b  = (const float*)d_in[15];
    const float* e2w1 = (const float*)d_in[16];
    const float* e2b1 = (const float*)d_in[17];
    const float* e2w2 = (const float*)d_in[18];
    const float* e2b2 = (const float*)d_in[19];
    const float* e2wx = (const float*)d_in[20];
    const float* e2bx = (const float*)d_in[21];
    const float* e2wo = (const float*)d_in[22];
    const float* e2bo = (const float*)d_in[23];
    const float* a2w  = (const float*)d_in[24];
    const float* a2b  = (const float*)d_in[25];
    const float* g1w  = (const float*)d_in[26];
    const float* g1b  = (const float*)d_in[27];
    const float* g2w  = (const float*)d_in[28];
    const float* g2b  = (const float*)d_in[29];
    const float* t1w  = (const float*)d_in[30];
    const float* t1b  = (const float*)d_in[31];
    const float* t2w  = (const float*)d_in[32];
    const float* t2b  = (const float*)d_in[33];
    const float* t3w  = (const float*)d_in[34];
    const float* t3b  = (const float*)d_in[35];

    float* ws = (float*)d_ws;
    const size_t BIG = 41943040;   // 327680 * 128
    float* R0    = ws;
    float* R1    = R0 + BIG;
    float* R2    = R1 + BIG;
    float* STYLE = R2 + BIG;           // 32768*128
    float* X1    = STYLE + 4194304;    // 32768*64
    float* X2    = X1 + 2097152;       // 32768*128
    int*   IDX   = (int*)(X2 + 4194304);   // 327680 ints
    float* MEAN  = (float*)(IDX + 327680); // 4096
    float* VAR   = MEAN + 4096;            // 4096
    float* FG    = VAR + 4096;             // 2048
    float* H1    = FG + 2048;              // 2048
    float* H2    = H1 + 2048;              // 8192
    float* WOT   = H2 + 8192;              // 163840

    const int M1 = BB * NN;        // 32768
    const int M2 = BB * NN * KK;   // 327680
    const int TPB = 256;
    auto blocks = [](size_t total) { return (unsigned)((total + 255) / 256); };

    auto gemm = [&](const float* A, int lda, int M, int Kc, const float* W,
                    const float* bias, float* Out, int ldo, int O, float slope) {
        dim3 grid((unsigned)((M + 63) / 64), (unsigned)((O + 63) / 64));
        hipLaunchKernelGGL(gemm_wmma, grid, dim3(128), 0, stream,
                           A, lda, M, Kc, W, Kc, bias, Out, ldo, O, slope);
    };

    // ---- style MLP: (x|z) -> 128 -> 128, lrelu 0.01 ----
    hipLaunchKernelGGL(build_style_in, dim3(blocks((size_t)M1 * 131)), dim3(TPB), 0, stream, x, z, R0);
    gemm(R0, 131, M1, 131, hw1, hb1, R1, 128, 128, 0.01f);
    gemm(R1, 128, M1, 128, hw2, hb2, STYLE, 128, 128, 0.01f);

    // ---- edge block 1 (C=3 -> 64) ----
    hipLaunchKernelGGL((knn_kernel<3, 2048>), dim3(BB * (NN / 256)), dim3(TPB), 0, stream, x, IDX);
    hipLaunchKernelGGL(build_ee, dim3(blocks((size_t)M2 * 6)), dim3(TPB), 0, stream, x, IDX, R0, 3, (size_t)M2 * 6);
    gemm(R0 + 3, 6, M2, 3, e1w1, e1b1, R1, 32, 32, 1.0f);                    // c2 w1
    hipLaunchKernelGGL(colstats, dim3(32), dim3(TPB), 0, stream, R1, 32, (size_t)M2, MEAN, VAR);
    hipLaunchKernelGGL(bn_apply, dim3(blocks((size_t)M2 * 32)), dim3(TPB), 0, stream, R1, 32, (size_t)M2 * 32, MEAN, VAR, 0.01f);
    gemm(R1, 32, M2, 32, e1w2, e1b2, R2, 64, 64, 1.0f);                      // c2 w2
    hipLaunchKernelGGL(colstats, dim3(64), dim3(TPB), 0, stream, R2, 64, (size_t)M2, MEAN, VAR);
    hipLaunchKernelGGL(bn_apply, dim3(blocks((size_t)M2 * 64)), dim3(TPB), 0, stream, R2, 64, (size_t)M2 * 64, MEAN, VAR, 0.01f);
    hipLaunchKernelGGL(softmax_k, dim3(blocks((size_t)M1 * 64)), dim3(TPB), 0, stream, R2, 64, (size_t)M1 * 64);
    gemm(R0, 6, M2, 6, e1wx, e1bx, R1, 64, 64, 1.0f);                        // c2 wx
    hipLaunchKernelGGL(colstats, dim3(64), dim3(TPB), 0, stream, R1, 64, (size_t)M2, MEAN, VAR);
    hipLaunchKernelGGL(bn_apply, dim3(blocks((size_t)M2 * 64)), dim3(TPB), 0, stream, R1, 64, (size_t)M2 * 64, MEAN, VAR, 0.01f);
    hipLaunchKernelGGL(mul_inplace, dim3(blocks((size_t)M2 * 64)), dim3(TPB), 0, stream, R1, R2, (size_t)M2 * 64);
    hipLaunchKernelGGL(transpose_wo, dim3(blocks((size_t)64 * 64 * KK)), dim3(TPB), 0, stream, e1wo, WOT, 64, 64);
    gemm(R1, 640, M1, 640, WOT, e1bo, X1, 64, 64, 0.2f);                     // aggregate + lrelu 0.2
    gemm(STYLE, 128, M1, 128, a1w, a1b, R2, 128, 128, 1.0f);                 // adain scale/shift
    hipLaunchKernelGGL(inorm_stats, dim3(BB * 64), dim3(TPB), 0, stream, X1, 64, MEAN, VAR);
    hipLaunchKernelGGL(adain_apply, dim3(blocks((size_t)M1 * 64)), dim3(TPB), 0, stream, X1, R2, 64, MEAN, VAR);

    // ---- edge block 2 (C=64 -> 128) ----
    hipLaunchKernelGGL((knn_kernel<64, 128>), dim3(BB * (NN / 256)), dim3(TPB), 0, stream, X1, IDX);
    hipLaunchKernelGGL(build_ee, dim3(blocks((size_t)M2 * 128)), dim3(TPB), 0, stream, X1, IDX, R0, 64, (size_t)M2 * 128);
    gemm(R0 + 64, 128, M2, 64, e2w1, e2b1, R1, 64, 64, 1.0f);
    hipLaunchKernelGGL(colstats, dim3(64), dim3(TPB), 0, stream, R1, 64, (size_t)M2, MEAN, VAR);
    hipLaunchKernelGGL(bn_apply, dim3(blocks((size_t)M2 * 64)), dim3(TPB), 0, stream, R1, 64, (size_t)M2 * 64, MEAN, VAR, 0.01f);
    gemm(R1, 64, M2, 64, e2w2, e2b2, R2, 128, 128, 1.0f);
    hipLaunchKernelGGL(colstats, dim3(128), dim3(TPB), 0, stream, R2, 128, (size_t)M2, MEAN, VAR);
    hipLaunchKernelGGL(bn_apply, dim3(blocks((size_t)M2 * 128)), dim3(TPB), 0, stream, R2, 128, (size_t)M2 * 128, MEAN, VAR, 0.01f);
    hipLaunchKernelGGL(softmax_k, dim3(blocks((size_t)M1 * 128)), dim3(TPB), 0, stream, R2, 128, (size_t)M1 * 128);
    gemm(R0, 128, M2, 128, e2wx, e2bx, R1, 128, 128, 1.0f);
    hipLaunchKernelGGL(colstats, dim3(128), dim3(TPB), 0, stream, R1, 128, (size_t)M2, MEAN, VAR);
    hipLaunchKernelGGL(bn_apply, dim3(blocks((size_t)M2 * 128)), dim3(TPB), 0, stream, R1, 128, (size_t)M2 * 128, MEAN, VAR, 0.01f);
    hipLaunchKernelGGL(mul_inplace, dim3(blocks((size_t)M2 * 128)), dim3(TPB), 0, stream, R1, R2, (size_t)M2 * 128);
    hipLaunchKernelGGL(transpose_wo, dim3(blocks((size_t)128 * 128 * KK)), dim3(TPB), 0, stream, e2wo, WOT, 128, 128);
    gemm(R1, 1280, M1, 1280, WOT, e2bo, X2, 128, 128, 0.2f);
    gemm(STYLE, 128, M1, 128, a2w, a2b, R2, 256, 256, 1.0f);
    hipLaunchKernelGGL(inorm_stats, dim3(BB * 128), dim3(TPB), 0, stream, X2, 128, MEAN, VAR);
    hipLaunchKernelGGL(adain_apply, dim3(blocks((size_t)M1 * 128)), dim3(TPB), 0, stream, X2, R2, 128, MEAN, VAR);

    // ---- global feature: maxpool -> fc(128) -> bn1 -> fc(512) -> bn1 ----
    hipLaunchKernelGGL(rowmax, dim3(blocks(BB * 128)), dim3(TPB), 0, stream, X2, 128, FG);
    gemm(FG, 128, BB, 128, g1w, g1b, H1, 128, 128, 1.0f);
    hipLaunchKernelGGL(bn1_apply, dim3(1), dim3(128), 0, stream, H1, 128, 0.01f);
    gemm(H1, 128, BB, 128, g2w, g2b, H2, 512, 512, 1.0f);
    hipLaunchKernelGGL(bn1_apply, dim3(2), dim3(TPB), 0, stream, H2, 512, 0.01f);

    // ---- tail: concat(fg, x2) -> 256 -> 64 -> 3 ----
    hipLaunchKernelGGL(build_feat, dim3(blocks((size_t)M1 * 640)), dim3(TPB), 0, stream, H2, X2, R0);
    gemm(R0, 640, M1, 640, t1w, t1b, R1, 256, 256, 0.01f);
    gemm(R1, 256, M1, 256, t2w, t2b, R2, 64, 64, 0.01f);
    gemm(R2, 64, M1, 64, t3w, t3b, (float*)d_out, 3, 3, 1.0f);
}
